// GatGegn_27762668601923
// MI455X (gfx1250) — matrix operand order repaired and checked by reading, not compiled, not run on hardware
//
#include <hip/hip_runtime.h>

typedef __attribute__((ext_vector_type(16))) __bf16 v16bf;
typedef __attribute__((ext_vector_type(8)))  float  v8f;

#define DEV static __device__ __forceinline__

DEV unsigned short f2bf(float f) {
  unsigned int u = __float_as_uint(f);
  return (unsigned short)((u + 0x7fffu + ((u >> 16) & 1u)) >> 16);  // RNE
}
DEV float bf2f(unsigned short h) { return __uint_as_float(((unsigned int)h) << 16); }
DEV int   ordenc(float f) { int i = __float_as_int(f); return (i >= 0) ? i : (i ^ 0x7fffffff); }
DEV float orddec(int v)   { int i = (v >= 0) ? v : (v ^ 0x7fffffff); return __int_as_float(i); }

// ---------------- generic fill ----------------
__global__ void k_fill32(unsigned int* __restrict__ p, unsigned int v, long long count) {
  long long stride = (long long)gridDim.x * blockDim.x;
  for (long long i = (long long)blockIdx.x * blockDim.x + threadIdx.x; i < count; i += stride)
    p[i] = v;
}

// ------------- fp32 -> bf16 activation, with K padding -------------
__global__ void k_cvt_act(const float* __restrict__ in, unsigned short* __restrict__ out,
                          long long rows, int cin, int cpad) {
  long long total = rows * cpad;
  long long stride = (long long)gridDim.x * blockDim.x;
  for (long long i = (long long)blockIdx.x * blockDim.x + threadIdx.x; i < total; i += stride) {
    long long r = i / cpad; int c = (int)(i - r * cpad);
    out[i] = (c < cin) ? f2bf(in[r * cin + c]) : (unsigned short)0;
  }
}

// ------------- fp32 weight [K x N] -> bf16 pre-swizzled B fragments -------------
// Fragment block (1024 B) = 32 lanes x 16 bf16; lane l, element e of k-tile kt,
// n-tile nt holds W[kt*32 + (l>>4)*16 + e, nt*16 + (l&15)].  In the GEMM each lane
// then reads its 16 elements with two contiguous 16-B loads (coalesced per wave).
__global__ void k_pack_w(const float* __restrict__ in, unsigned short* __restrict__ out,
                         int K, int Kpad, int N) {
  long long total = (long long)Kpad * N;
  long long stride = (long long)gridDim.x * blockDim.x;
  const int ntiles = N >> 4;
  for (long long i = (long long)blockIdx.x * blockDim.x + threadIdx.x; i < total; i += stride) {
    int e    = (int)(i & 15);
    int lane = (int)((i >> 4) & 31);
    long long rest = i >> 9;
    int nt = (int)(rest % ntiles);
    int kt = (int)(rest / ntiles);
    int k = kt * 32 + (lane >> 4) * 16 + e;
    int n = nt * 16 + (lane & 15);
    out[i] = (k < K) ? f2bf(in[(long long)k * N + n]) : (unsigned short)0;
  }
}

// ------------- WMMA bf16 GEMM: C[rows x N] = A[rows x K] * Wpacked -------------
// block = 256 threads = 8 waves; each wave: 16 rows x (16*NT) cols -> 128 rows/block.
// rows % 128 == 0, K % 32 == 0, N % (16*NT) == 0.
// Double-buffered K pipeline: fragments for step i+1 are issued before the WMMAs
// of step i so the s_wait before each WMMA group covers latency already overlapped.
template <int NT>
__global__ __launch_bounds__(256)
void k_gemm_bf16(const unsigned short* __restrict__ A, int lda,
                 const unsigned short* __restrict__ Wp, int N, int K,
                 float* __restrict__ C) {
  const int lane = threadIdx.x & 31;
  const int half = lane >> 4;        // K-half selector (contiguous k = 16*half + s, same in A and B)
  const int ml   = lane & 15;        // row (A) / col (C) within tile
  const long long m0 = ((long long)blockIdx.x * 8 + (threadIdx.x >> 5)) * 16;
  const int n0 = blockIdx.y * (NT * 16);
  const int ntiles = N >> 4;

  v8f acc[NT];
#pragma unroll
  for (int t = 0; t < NT; ++t)
#pragma unroll
    for (int i = 0; i < 8; ++i) acc[t][i] = 0.f;

  union AB { uint4 q[2]; v16bf v; };
  const unsigned short* arow  = A + (m0 + ml) * (long long)lda;
  const unsigned short* wlane = Wp + ((long long)(n0 >> 4) << 9) + lane * 16;
  const long long wkstep = (long long)ntiles << 9;  // elements per 32-K of packed W

  AB a0, a1; AB b0[NT], b1[NT];
#define LOAD_A(dstA, kk)                                              \
  do {                                                                \
    (dstA).q[0] = *(const uint4*)(arow + (kk) + half * 16);           \
    (dstA).q[1] = *(const uint4*)(arow + (kk) + half * 16 + 8);       \
  } while (0)
#define LOAD_B(dstB, kk)                                              \
  do {                                                                \
    const unsigned short* wb = wlane + (long long)((kk) >> 5) * wkstep; \
    _Pragma("unroll")                                                 \
    for (int t = 0; t < NT; ++t) {                                    \
      (dstB)[t].q[0] = *(const uint4*)(wb + ((long long)t << 9));     \
      (dstB)[t].q[1] = *(const uint4*)(wb + ((long long)t << 9) + 8); \
    }                                                                 \
  } while (0)
#define DO_MMA(srcA, srcB)                                            \
  do {                                                                \
    _Pragma("unroll")                                                 \
    for (int t = 0; t < NT; ++t)                                      \
      acc[t] = __builtin_amdgcn_wmma_f32_16x16x32_bf16(               \
          false, (srcA).v, false, (srcB)[t].v, (short)0, acc[t],      \
          false, false);                                              \
  } while (0)

  const int n32 = K >> 5;
  LOAD_A(a0, 0); LOAD_B(b0, 0);
  int it = 0;
  for (; it + 2 <= n32; it += 2) {
    LOAD_A(a1, (it + 1) * 32); LOAD_B(b1, (it + 1) * 32);
    DO_MMA(a0, b0);
    if (it + 2 < n32) { LOAD_A(a0, (it + 2) * 32); LOAD_B(b0, (it + 2) * 32); }
    DO_MMA(a1, b1);
  }
  if (it < n32) DO_MMA(a0, b0);   // odd K/32 tail (last tile already in buffer 0)
#undef LOAD_A
#undef LOAD_B
#undef DO_MMA

  // D layout: VGPR r -> (M = r + 8*half, N = ml)
#pragma unroll
  for (int t = 0; t < NT; ++t) {
    float* cp = C + (m0 + half * 8) * (long long)N + n0 + t * 16 + ml;
#pragma unroll
    for (int r = 0; r < 8; ++r) cp[(long long)r * N] = acc[t][r];
  }
}

// ------------- edge list (reference-faithful: + per-sample offset b, + self loops) -------------
__global__ void k_build_edges(const int* __restrict__ ei, int* __restrict__ src,
                              int* __restrict__ dst, long long E0, long long E) {
  long long stride = (long long)gridDim.x * blockDim.x;
  for (long long i = (long long)blockIdx.x * blockDim.x + threadIdx.x; i < E; i += stride) {
    if (i < E0) {
      long long b = i / 48; int e = (int)(i - b * 48);
      src[i] = ei[b * 96 + e]      + (int)b;
      dst[i] = ei[b * 96 + 48 + e] + (int)b;
    } else {
      int v = (int)(i - E0);
      src[i] = v; dst[i] = v;
    }
  }
}

// ------------- per-node attention scalars: a_src = h.att_src, a_dst = h.att_dst -------------
__global__ __launch_bounds__(256)
void k_attn_dots(const float* __restrict__ h, const float* __restrict__ vs,
                 const float* __restrict__ vd, float* __restrict__ os,
                 float* __restrict__ od, long long nn, int C) {
  const int lane = threadIdx.x & 31;
  long long w  = (long long)blockIdx.x * (blockDim.x >> 5) + (threadIdx.x >> 5);
  long long nw = (long long)gridDim.x * (blockDim.x >> 5);
  for (long long node = w; node < nn; node += nw) {
    const float* hp = h + node * C;
    float s0 = 0.f, s1 = 0.f;
    for (int c = lane * 4; c < C; c += 128) {
      float4 hv = *(const float4*)(hp + c);
      float4 a  = *(const float4*)(vs + c);
      float4 b  = *(const float4*)(vd + c);
      s0 += hv.x*a.x + hv.y*a.y + hv.z*a.z + hv.w*a.w;
      s1 += hv.x*b.x + hv.y*b.y + hv.z*b.z + hv.w*b.w;
    }
#pragma unroll
    for (int o = 16; o; o >>= 1) { s0 += __shfl_xor(s0, o, 32); s1 += __shfl_xor(s1, o, 32); }
    if (lane == 0) { os[node] = s0; od[node] = s1; }
  }
}

// ------------- leaky-relu logits + per-dst max (ordered-int atomicMax) -------------
__global__ void k_edge_logit_max(const float* __restrict__ as, const float* __restrict__ ad,
                                 const int* __restrict__ src, const int* __restrict__ dst,
                                 float* __restrict__ elog, int* __restrict__ mord, long long E) {
  long long stride = (long long)gridDim.x * blockDim.x;
  for (long long i = (long long)blockIdx.x * blockDim.x + threadIdx.x; i < E; i += stride) {
    float l = as[src[i]] + ad[dst[i]];
    l = (l >= 0.f) ? l : 0.2f * l;
    elog[i] = l;
    atomicMax(&mord[dst[i]], ordenc(l));
  }
}

__global__ void k_decode_max(int* __restrict__ mord, long long nn) {
  float* mf = (float*)mord;
  long long stride = (long long)gridDim.x * blockDim.x;
  for (long long i = (long long)blockIdx.x * blockDim.x + threadIdx.x; i < nn; i += stride) {
    int v = mord[i];
    mf[i] = orddec(v);
  }
}

__global__ void k_edge_exp_sum(float* __restrict__ elog, const float* __restrict__ mf,
                               const int* __restrict__ dst, float* __restrict__ ssum, long long E) {
  long long stride = (long long)gridDim.x * blockDim.x;
  for (long long i = (long long)blockIdx.x * blockDim.x + threadIdx.x; i < E; i += stride) {
    float e = __expf(elog[i] - mf[dst[i]]);
    elog[i] = e;
    atomicAdd(&ssum[dst[i]], e);
  }
}

// ------------- out[dst] += (e/s[dst]) * h[src], wave per edge -------------
__global__ __launch_bounds__(256)
void k_edge_aggr(const float* __restrict__ h, const float* __restrict__ elog,
                 const float* __restrict__ ssum, const int* __restrict__ src,
                 const int* __restrict__ dst, float* __restrict__ outp,
                 long long E, int C) {
  const int lane = threadIdx.x & 31;
  long long w  = (long long)blockIdx.x * (blockDim.x >> 5) + (threadIdx.x >> 5);
  long long nw = (long long)gridDim.x * (blockDim.x >> 5);
  for (long long j = w; j < E; j += nw) {
    int sN = src[j], dN = dst[j];
    float alpha = elog[j] / ssum[dN];
    const float* hp = h + (long long)sN * C;
    float* op = outp + (long long)dN * C;
    for (int c = lane * 4; c < C; c += 128) {
      float4 hv = *(const float4*)(hp + c);
      atomicAdd(op + c + 0, alpha * hv.x);
      atomicAdd(op + c + 1, alpha * hv.y);
      atomicAdd(op + c + 2, alpha * hv.z);
      atomicAdd(op + c + 3, alpha * hv.w);
    }
  }
}

// ------------- BN pass 1: per-channel sum / sumsq  (C in {32,64,128,256}) -------------
__global__ __launch_bounds__(256)
void k_bn_reduce(const float* __restrict__ x, long long rows, int C, float* __restrict__ acc) {
  const int c  = threadIdx.x & (C - 1);
  const int rg = threadIdx.x / C;
  const int nrg = 256 / C;
  float s = 0.f, q = 0.f;
  for (long long r = (long long)blockIdx.x * nrg + rg; r < rows; r += (long long)gridDim.x * nrg) {
    float v = x[r * C + c];
    s += v; q += v * v;
  }
  __shared__ float ls[256], lq[256];
  ls[threadIdx.x] = s; lq[threadIdx.x] = q;
  __syncthreads();
  if (rg == 0) {
    for (int g = 1; g < nrg; ++g) { s += ls[g * C + c]; q += lq[g * C + c]; }
    atomicAdd(&acc[c], s);
    atomicAdd(&acc[C + c], q);
  }
}

// ------------- BN pass 2 + PReLU + f32->bf16, optional row-gather (center nodes) -------------
__global__ void k_bn_apply(const float* __restrict__ x, const float* __restrict__ acc,
                           const float* __restrict__ gamma, const float* __restrict__ beta,
                           const float* __restrict__ prelu, unsigned short* __restrict__ outp,
                           long long outRows, int C, int ldo, int colOff,
                           int rstep, int roff, float invN) {
  const float a = prelu[0];
  long long total = outRows * C;
  long long stride = (long long)gridDim.x * blockDim.x;
  for (long long i = (long long)blockIdx.x * blockDim.x + threadIdx.x; i < total; i += stride) {
    long long r = i / C; int c = (int)(i - r * C);
    float mean = acc[c] * invN;
    float var  = acc[C + c] * invN - mean * mean;   // biased var, matches x.var(0)
    float v = (x[(r * (long long)rstep + roff) * (long long)C + c] - mean)
              * rsqrtf(var + 1e-5f) * gamma[c] + beta[c];
    v = (v >= 0.f) ? v : a * v;
    outp[r * (long long)ldo + colOff + c] = f2bf(v);
  }
}

// ------------- final head: out[r] = act[r,:].Wf + bf -------------
__global__ __launch_bounds__(256)
void k_final_gemv(const unsigned short* __restrict__ act, const float* __restrict__ W,
                  const float* __restrict__ b, float* __restrict__ outp,
                  long long rows, int C) {
  const int lane = threadIdx.x & 31;
  long long w  = (long long)blockIdx.x * (blockDim.x >> 5) + (threadIdx.x >> 5);
  long long nw = (long long)gridDim.x * (blockDim.x >> 5);
  for (long long r = w; r < rows; r += nw) {
    float s = 0.f;
    for (int c = lane; c < C; c += 32) s += bf2f(act[r * C + c]) * W[c];
#pragma unroll
    for (int o = 16; o; o >>= 1) s += __shfl_xor(s, o, 32);
    if (lane == 0) outp[r] = s + b[0];
  }
}

extern "C" void kernel_launch(void* const* d_in, const int* in_sizes, int n_in,
                              void* d_out, int out_size, void* d_ws, size_t ws_size,
                              hipStream_t stream) {
  (void)in_sizes; (void)n_in; (void)out_size;
  const long long Bq = 16384, GN = 25;
  const long long NN = Bq * GN;        // 409600 nodes
  const long long E0 = Bq * 48;        // real edges
  const long long E  = E0 + NN;        // + self loops

  // ---- inputs (jax pytree flatten order: dict keys sorted) ----
  const float* o_x    = (const float*)d_in[0];
  const float* d_x    = (const float*)d_in[1];
  const float* edge_x = (const float*)d_in[2];
  const float* ext_x  = (const float*)d_in[3];
  const int*   eidx   = (const int*)d_in[4];
  // grid GAT layers: W, att_dst, att_src, b, beta, gamma, prelu  (biases cancel under BN)
  const float* gW[2]    = {(const float*)d_in[5],  (const float*)d_in[12]};
  const float* gAdst[2] = {(const float*)d_in[6],  (const float*)d_in[13]};
  const float* gAsrc[2] = {(const float*)d_in[7],  (const float*)d_in[14]};
  const float* gBeta[2] = {(const float*)d_in[9],  (const float*)d_in[16]};
  const float* gGam[2]  = {(const float*)d_in[10], (const float*)d_in[17]};
  const float* gPre[2]  = {(const float*)d_in[11], (const float*)d_in[18]};
  // edge MLP: W, b, beta, gamma, prelu
  const float* eW[2]    = {(const float*)d_in[19], (const float*)d_in[24]};
  const float* eBeta[2] = {(const float*)d_in[21], (const float*)d_in[26]};
  const float* eGam[2]  = {(const float*)d_in[22], (const float*)d_in[27]};
  const float* ePre[2]  = {(const float*)d_in[23], (const float*)d_in[28]};
  // external MLP
  const float* xW[2]    = {(const float*)d_in[29], (const float*)d_in[34]};
  const float* xBeta[2] = {(const float*)d_in[31], (const float*)d_in[36]};
  const float* xGam[2]  = {(const float*)d_in[32], (const float*)d_in[37]};
  const float* xPre[2]  = {(const float*)d_in[33], (const float*)d_in[38]};
  // out params: "final" (W,b) then "hidden" (W,b,beta,gamma,prelu)
  const float* fW    = (const float*)d_in[39];
  const float* fB    = (const float*)d_in[40];
  const float* hW    = (const float*)d_in[41];
  const float* hBeta = (const float*)d_in[43];
  const float* hGam  = (const float*)d_in[44];
  const float* hPre  = (const float*)d_in[45];
  float* out = (float*)d_out;

  // ---- workspace carve ----
  char* base = (char*)d_ws; size_t woff = 0;
  auto carve = [&](size_t bytes) -> void* {
    void* r = base + woff; woff = (woff + bytes + 255) & ~(size_t)255; return r;
  };
  int*   src   = (int*)carve(E * 4);
  int*   dst   = (int*)carve(E * 4);
  float* elog  = (float*)carve(E * 4);
  int*   mord  = (int*)carve(NN * 4);
  float* ssum  = (float*)carve(NN * 4);
  float* asv   = (float*)carve(NN * 4);
  float* adv   = (float*)carve(NN * 4);
  float* bnacc = (float*)carve(512 * 4);
  unsigned short* wbf  = (unsigned short*)carve(352 * 256 * 2);
  unsigned short* xbf  = (unsigned short*)carve(NN * 128 * 2);
  float* hbuf  = (float*)carve(NN * 128 * 4);
  float* aggr  = (float*)carve(NN * 128 * 4);
  unsigned short* totb = (unsigned short*)carve(Bq * 352 * 2);
  if (woff > ws_size) return;   // insufficient scratch
  // MLP phase reuses the (then-idle) GAT buffers
  float* mlpH = hbuf;                                        // up to B x 256 f32
  unsigned short* mlpIn  = (unsigned short*)aggr;            // up to B x 64 bf16
  unsigned short* mlpAct = (unsigned short*)((char*)aggr + (4u << 20)); // up to B x 256 bf16

  auto gs = [](long long total, int per) -> unsigned {
    long long b = (total + per - 1) / per;
    if (b > 8192) b = 8192; if (b < 1) b = 1; return (unsigned)b;
  };

  // ---- edges (built once, shared by both stacks and both layers) ----
  k_build_edges<<<gs(E, 256), 256, 0, stream>>>(eidx, src, dst, E0, E);

  // ---- one GAT layer: GEMM -> attention -> softmax scatter -> BN stats ----
  auto gat_layer = [&](const unsigned short* xin, int K, int layer) {
    const int C = 128;
    k_pack_w<<<gs((long long)K * C, 256), 256, 0, stream>>>(gW[layer], wbf, K, K, C);
    dim3 g((unsigned)(NN / 128), (unsigned)(C / 64));
    k_gemm_bf16<4><<<g, 256, 0, stream>>>(xin, K, wbf, C, K, hbuf);
    k_attn_dots<<<gs(NN * 32, 256), 256, 0, stream>>>(hbuf, gAsrc[layer], gAdst[layer], asv, adv, NN, C);
    k_fill32<<<gs(NN, 256), 256, 0, stream>>>((unsigned int*)mord, 0x80000000u, NN);
    k_fill32<<<gs(NN, 256), 256, 0, stream>>>((unsigned int*)ssum, 0u, NN);
    k_fill32<<<gs(NN * C, 256), 256, 0, stream>>>((unsigned int*)aggr, 0u, NN * C);
    k_edge_logit_max<<<gs(E, 256), 256, 0, stream>>>(asv, adv, src, dst, elog, mord, E);
    k_decode_max<<<gs(NN, 256), 256, 0, stream>>>(mord, NN);
    k_edge_exp_sum<<<gs(E, 256), 256, 0, stream>>>(elog, (const float*)mord, dst, ssum, E);
    k_edge_aggr<<<gs(E * 32, 256), 256, 0, stream>>>(hbuf, elog, ssum, src, dst, aggr, E, C);
    k_fill32<<<1, 256, 0, stream>>>((unsigned int*)bnacc, 0u, 512);
    k_bn_reduce<<<512, 256, 0, stream>>>(aggr, NN, C, bnacc);
  };

  auto run_stack = [&](const float* x_in, int colOff) {
    // layer 0: 64 -> 128, full-node BN+PReLU into bf16 for next layer
    k_cvt_act<<<gs(NN * 64, 256), 256, 0, stream>>>(x_in, xbf, NN, 64, 64);
    gat_layer(xbf, 64, 0);
    k_bn_apply<<<gs(NN * 128, 256), 256, 0, stream>>>(aggr, bnacc, gGam[0], gBeta[0], gPre[0],
        xbf, NN, 128, 128, 0, 1, 0, 1.f / (float)NN);
    // layer 1: 128 -> 128, BN stats over all nodes, apply only on center nodes -> concat buffer
    gat_layer(xbf, 128, 1);
    k_bn_apply<<<gs(Bq * 128, 256), 256, 0, stream>>>(aggr, bnacc, gGam[1], gBeta[1], gPre[1],
        totb, Bq, 128, 352, colOff, (int)GN, (int)(GN / 2), 1.f / (float)NN);
  };
  run_stack(o_x, 0);
  run_stack(d_x, 128);

  // ---- dense MLP layer: WMMA GEMM -> BN -> PReLU -> bf16 ----
  auto mlp_layer = [&](const unsigned short* inbf, int Kpad, const float* W, int Kreal, int N,
                       const float* gam, const float* bet, const float* pre,
                       unsigned short* outp, int ldo, int colOff, int NT) {
    k_pack_w<<<gs((long long)Kpad * N, 256), 256, 0, stream>>>(W, wbf, Kreal, Kpad, N);
    dim3 g((unsigned)(Bq / 128), (unsigned)(N / (16 * NT)));
    if (NT == 4) k_gemm_bf16<4><<<g, 256, 0, stream>>>(inbf, Kpad, wbf, N, Kpad, mlpH);
    else         k_gemm_bf16<2><<<g, 256, 0, stream>>>(inbf, Kpad, wbf, N, Kpad, mlpH);
    k_fill32<<<1, 256, 0, stream>>>((unsigned int*)bnacc, 0u, 512);
    k_bn_reduce<<<512, 256, 0, stream>>>(mlpH, Bq, N, bnacc);
    k_bn_apply<<<gs(Bq * N, 256), 256, 0, stream>>>(mlpH, bnacc, gam, bet, pre,
        outp, Bq, N, ldo, colOff, 1, 0, 1.f / (float)Bq);
  };

  // edge branch: 32 -> 64 -> 64, final into concat cols [256,320)
  k_cvt_act<<<gs(Bq * 32, 256), 256, 0, stream>>>(edge_x, mlpIn, Bq, 32, 32);
  mlp_layer(mlpIn, 32, eW[0], 32, 64, eGam[0], eBeta[0], ePre[0], mlpAct, 64, 0, 4);
  mlp_layer(mlpAct, 64, eW[1], 64, 64, eGam[1], eBeta[1], ePre[1], totb, 352, 256, 4);

  // external branch: 16(pad 32) -> 32 -> 32, final into concat cols [320,352)
  k_cvt_act<<<gs(Bq * 32, 256), 256, 0, stream>>>(ext_x, mlpIn, Bq, 16, 32);
  mlp_layer(mlpIn, 32, xW[0], 16, 32, xGam[0], xBeta[0], xPre[0], mlpAct, 32, 0, 2);
  mlp_layer(mlpAct, 32, xW[1], 32, 32, xGam[1], xBeta[1], xPre[1], totb, 352, 320, 2);

  // hidden: 352 -> 256
  mlp_layer(totb, 352, hW, 352, 256, hGam, hBeta, hPre, mlpAct, 256, 0, 4);

  // final head: 256 -> 1 (bias applied here; this one does NOT feed a BN)
  k_final_gemv<<<gs(Bq * 32, 256), 256, 0, stream>>>(mlpAct, fW, fB, out, Bq, 256);
}
